// QuantConv2d_33844342292937
// MI455X (gfx1250) — 8-bit matrix operand order repaired and checked by reading, compiled, not run on hardware
//
#include <hip/hip_runtime.h>
#include <hip/hip_bf16.h>

typedef __attribute__((ext_vector_type(8))) int v8i;

#define H 224
#define W 224
#define CIN 32
#define COUT 32
#define RSTRIDE 228
#define CONV_THREADS 448


__global__ void init_scales_kernel(unsigned* scales) {
    if (threadIdx.x < 2) scales[threadIdx.x] = 0u;
}

__global__ void absmax_kernel(const float* __restrict__ x, int n, unsigned* target) {
    __shared__ unsigned red[256];
    float m = 0.0f;
    for (int i = blockIdx.x * blockDim.x + threadIdx.x; i < n;
         i += gridDim.x * blockDim.x)
        m = fmaxf(m, fabsf(x[i]));
    red[threadIdx.x] = __float_as_uint(m);
    __syncthreads();
    for (int s = 128; s > 0; s >>= 1) {
        if ((int)threadIdx.x < s)
            red[threadIdx.x] = max(red[threadIdx.x], red[threadIdx.x + s]);
        __syncthreads();
    }
    if (threadIdx.x == 0) atomicMax(target, red[0]);
}

__global__ void pack_weight_kernel(const float* __restrict__ w,
                                   int* __restrict__ wpack,
                                   const unsigned* __restrict__ scales) {
    int tid = blockIdx.x * blockDim.x + threadIdx.x;
    if (tid >= 2 * 5 * 32 * 8) return;
    const float s = __uint_as_float(scales[1]) * (1.0f / 127.0f);
    const int v = tid & 7;
    int r = tid >> 3;
    const int lane = r & 31; r >>= 5;
    const int kt = r % 5;
    const int mtile = r / 5;
    const int half = lane >> 4;
    const int co = mtile * 16 + (lane & 15);
    unsigned dw = 0;
#pragma unroll
    for (int j = 0; j < 4; ++j) {
        const int klocal = ((v >= 4) ? 32 : 0) + half * 16
                         + ((v & 3) * 4) + j;
        const int k = kt * 64 + klocal;
        const int tap = k >> 5;
        const int c = k & 31;
        int qb = 0;
        if (tap < 9) {
            float q = rintf(w[(co * CIN + c) * 9 + tap] / s);
            q = fminf(fmaxf(q, -127.0f), 127.0f);
            qb = (int)q;
        }
        dw |= ((unsigned)(qb & 0xFF)) << (8 * j);
    }
    wpack[tid] = dw;
}

__global__ void quantize_x_kernel(const float* __restrict__ x,
                                  unsigned* __restrict__ xq,
                                  const unsigned* __restrict__ scales) {
    const int total = 8 * H * W * 8;
    int tid = blockIdx.x * blockDim.x + threadIdx.x;
    if (tid >= total) return;
    const float s = __uint_as_float(scales[0]) * (1.0f / 127.0f);
    const int cg  = tid & 7;
    const int pix = tid >> 3;
    const int wcol = pix % W;
    const int t    = pix / W;
    const int hh   = t % H;
    const int b    = t / H;
    const int c0   = cg * 4;
    const int base = ((b * CIN + c0) * H + hh) * W + wcol;
    unsigned dw = 0;
#pragma unroll
    for (int j = 0; j < 4; ++j) {
        float q = rintf(x[base + j * H * W] / s);
        q = fminf(fmaxf(q, -127.0f), 127.0f);
        dw |= ((unsigned)(((int)q) & 0xFF)) << (8 * j);
    }
    xq[tid] = dw;
}

__global__ __launch_bounds__(CONV_THREADS) void conv_wmma_kernel(
    const unsigned* __restrict__ xq,
    const int* __restrict__ wpack,
    const unsigned* __restrict__ scales,
    const float* __restrict__ bias,
    float* __restrict__ out) {
    __shared__ __align__(16) unsigned ldsu[3 * RSTRIDE * 8];

    const int bh = blockIdx.x;
    const int b  = bh / H;
    const int h  = bh % H;
    const int tid = threadIdx.x;

    for (int idx = tid; idx < 3 * 226 * 8; idx += CONV_THREADS) {
        const int r   = idx / (226 * 8);
        const int rem = idx - r * (226 * 8);
        const int p   = rem >> 3;
        const int d   = rem & 7;
        const int gh  = h - 1 + r;
        const int gw  = p - 1;
        unsigned val = 0u;
        if ((unsigned)gh < (unsigned)H && (unsigned)gw < (unsigned)W)
            val = xq[((b * H + gh) * W + gw) * 8 + d];
        ldsu[(r * RSTRIDE + p) * 8 + d] = val;
    }
    __syncthreads();

    const int wave = tid >> 5;
    const int lane = tid & 31;
    const int half = lane >> 4;
    const int col0 = wave * 16 + (lane & 15);

    v8i acc0 = {0, 0, 0, 0, 0, 0, 0, 0};
    v8i acc1 = {0, 0, 0, 0, 0, 0, 0, 0};
    const int aBase = lane * 8;

#pragma unroll
    for (int kt = 0; kt < 5; ++kt) {
        const int tapA = 2 * kt, tapB = 2 * kt + 1;
        const int khA = tapA / 3, kwA = tapA % 3;
        const int4 lo = *(const int4*)&ldsu[((khA * RSTRIDE) + col0 + kwA) * 8 + half * 4];
        int4 hi = make_int4(0, 0, 0, 0);
        if (tapB < 9) {
            const int khB = tapB / 3, kwB = tapB % 3;
            hi = *(const int4*)&ldsu[((khB * RSTRIDE) + col0 + kwB) * 8 + half * 4];
        }
        v8i bfrag = {lo.x, lo.y, lo.z, lo.w, hi.x, hi.y, hi.z, hi.w};
        v8i a0 = *(const v8i*)&wpack[(0 * 5 + kt) * 256 + aBase];
        v8i a1 = *(const v8i*)&wpack[(1 * 5 + kt) * 256 + aBase];
        acc0 = __builtin_amdgcn_wmma_i32_16x16x64_iu8(true, a0, true, bfrag, acc0, false, false);
        acc1 = __builtin_amdgcn_wmma_i32_16x16x64_iu8(true, a1, true, bfrag, acc1, false, false);
    }

    const float sc = (__uint_as_float(scales[0]) * (1.0f / 127.0f)) *
                     (__uint_as_float(scales[1]) * (1.0f / 127.0f));

#pragma unroll
    for (int i = 0; i < 8; ++i) {
        const int m0 = i + 8 * half;
        {
            const int co = m0;
            out[((b * COUT + co) * H + h) * W + col0] = (float)acc0[i] * sc + bias[co];
        }
        {
            const int co = 16 + m0;
            out[((b * COUT + co) * H + h) * W + col0] = (float)acc1[i] * sc + bias[co];
        }
    }
}

extern "C" void kernel_launch(void* const* d_in, const int* in_sizes, int n_in,
                              void* d_out, int out_size, void* d_ws, size_t ws_size,
                              hipStream_t stream) {
    const float* x      = (const float*)d_in[0];
    const float* weight = (const float*)d_in[1];
    const float* bias   = (const float*)d_in[2];
    float* out = (float*)d_out;

    unsigned* scales = (unsigned*)d_ws;
    int*      wpack  = (int*)((char*)d_ws + 64);
    unsigned* xq     = (unsigned*)((char*)d_ws + 16384);

    const int n_x = 8 * CIN * H * W;
    const int n_w = COUT * CIN * 9;

    init_scales_kernel<<<1, 32, 0, stream>>>(scales);
    absmax_kernel<<<2048, 256, 0, stream>>>(x, n_x, scales + 0);
    absmax_kernel<<<36, 256, 0, stream>>>(weight, n_w, scales + 1);
    pack_weight_kernel<<<10, 256, 0, stream>>>(weight, wpack, scales);
    quantize_x_kernel<<<(8 * H * W * 8 + 255) / 256, 256, 0, stream>>>(x, xq, scales);
    conv_wmma_kernel<<<8 * H, CONV_THREADS, 0, stream>>>(xq, wpack, scales, bias, out);
}
